// VisionTransformer_29274497089574
// MI455X (gfx1250) — matrix operand order repaired and checked by reading, not compiled, not run on hardware
//
#include <hip/hip_runtime.h>

typedef __attribute__((ext_vector_type(16))) _Float16 v16h;
typedef __attribute__((ext_vector_type(8)))  _Float16 v8h;
typedef __attribute__((ext_vector_type(8)))  float    v8f;

#define CDIM 768
#define NTOK 196
#define NH   12
#define HD   64
#define MTOT (64 * 196)      // 12544 rows, = 98*128
#define MROWS 208            // 13*16 padded token rows
#define NPADK 224            // K padding (7*32) for attn@V
#define WMMA_F16(a, b, c) __builtin_amdgcn_wmma_f32_16x16x32_f16(false, (a), false, (b), (short)0, (c), false, false)

union FragH { v16h v; v8h h[2]; };

// A-fragment: lane l<16 holds row m=l, halfs 0..7 = K k..k+7, halfs 8..15 = K k+16..k+23;
// lanes 16..31 hold K k+8.. / k+24.. (ISA 7.12.2).  Base pointer already includes +lh*8.
__device__ __forceinline__ FragH load_frag_a(const _Float16* p) {
    FragH f; f.h[0] = *(const v8h*)p; f.h[1] = *(const v8h*)(p + 16); return f;
}
// B-fragment: lane holds output column, halfs 0..7 = K k..k+7, 8..15 = K k+16..k+23 (base includes +lh*8).
__device__ __forceinline__ FragH load_frag_b(const _Float16* p) {
    FragH f; f.h[0] = *(const v8h*)p; f.h[1] = *(const v8h*)(p + 16); return f;
}

// ---------------------------------------------------------------- convert f32 -> f16
__global__ void conv_f16_kernel(const float* __restrict__ src, _Float16* __restrict__ dst, int n) {
    for (int i = blockIdx.x * 256 + threadIdx.x; i < n; i += gridDim.x * 256)
        dst[i] = (_Float16)src[i];
}

// ---------------------------------------------------------------- layernorm (row = token), 256 thr, C=768
__global__ void __launch_bounds__(256) ln_kernel(const float* __restrict__ x,
                                                 const float* __restrict__ g,
                                                 const float* __restrict__ b,
                                                 _Float16* __restrict__ y) {
    int row = blockIdx.x;
    int tid = threadIdx.x;
    const float* xr = x + (size_t)row * CDIM;
    float v0 = xr[tid], v1 = xr[tid + 256], v2 = xr[tid + 512];
    float s  = v0 + v1 + v2;
    float s2 = v0 * v0 + v1 * v1 + v2 * v2;
    for (int m = 1; m < 32; m <<= 1) { s += __shfl_xor(s, m, 32); s2 += __shfl_xor(s2, m, 32); }
    __shared__ float ps[8], ps2[8];
    if ((tid & 31) == 0) { ps[tid >> 5] = s; ps2[tid >> 5] = s2; }
    __syncthreads();
    s = 0.f; s2 = 0.f;
    for (int i = 0; i < 8; i++) { s += ps[i]; s2 += ps2[i]; }
    float mean = s * (1.f / 768.f);
    float var  = s2 * (1.f / 768.f) - mean * mean;
    float inv  = rsqrtf(var + 1e-5f);
    _Float16* yr = y + (size_t)row * CDIM;
    yr[tid]       = (_Float16)((v0 - mean) * inv * g[tid]       + b[tid]);
    yr[tid + 256] = (_Float16)((v1 - mean) * inv * g[tid + 256] + b[tid + 256]);
    yr[tid + 512] = (_Float16)((v2 - mean) * inv * g[tid + 512] + b[tid + 512]);
}

// ---------------------------------------------------------------- WMMA GEMM: out = act(A @ W^T + bias)
// A: (M x K) f16 row-major. W: (Nout x K) f16 row-major.
// Block tile 128x128 (8 waves as 4Mx2N), wave tile 32x64, K-step 32.
// Software pipeline: loop body loads step k+1 unconditionally (pointer bump, fixed offsets),
// final step peeled into a loadless epilogue -> no per-iteration select/address rebuild.
// mode 0: f16 out;  mode 1: f16 out + exact GELU;  mode 2: f32 residual add into res.
__global__ void __launch_bounds__(256) gemm_kernel(const _Float16* __restrict__ A,
                                                   const _Float16* __restrict__ W,
                                                   const float* __restrict__ bias,
                                                   _Float16* __restrict__ out16,
                                                   float* __restrict__ res,
                                                   int K, int Nout, int mode) {
    const int lane = threadIdx.x & 31;
    const int wv   = threadIdx.x >> 5;
    const int l15  = lane & 15, lh = lane >> 4;
    const int wm   = wv & 3, wn = wv >> 2;
    const int m0   = blockIdx.y * 128 + wm * 32;
    const int n0   = blockIdx.x * 128 + wn * 64;

    const size_t ws16 = (size_t)16 * K;
    const _Float16* pa0 = A + (size_t)(m0 + l15)      * K + lh * 8;
    const _Float16* pa1 = A + (size_t)(m0 + 16 + l15) * K + lh * 8;
    const _Float16* pw0 = W + (size_t)(n0 + l15)      * K + lh * 8;
    const _Float16* pw1 = pw0 + ws16;
    const _Float16* pw2 = pw0 + 2 * ws16;
    const _Float16* pw3 = pw0 + 3 * ws16;

    v8f acc[8];
#pragma unroll
    for (int t = 0; t < 8; t++) acc[t] = (v8f){};

    // prologue: fragments for k-step 0
    FragH a0 = load_frag_a(pa0), a1 = load_frag_a(pa1);
    FragH b0 = load_frag_b(pw0), b1 = load_frag_b(pw1);
    FragH b2 = load_frag_b(pw2), b3 = load_frag_b(pw3);

#pragma unroll 2
    for (int kk = 0; kk < K - 32; kk += 32) {
        __builtin_prefetch(pw0 + 512, 0, 1);            // global_prefetch_b8 (speculative OK past end)
        // issue next-step loads before consuming current fragments
        FragH a0n = load_frag_a(pa0 + 32), a1n = load_frag_a(pa1 + 32);
        FragH b0n = load_frag_b(pw0 + 32), b1n = load_frag_b(pw1 + 32);
        FragH b2n = load_frag_b(pw2 + 32), b3n = load_frag_b(pw3 + 32);

        acc[0] = WMMA_F16(a0.v, b0.v, acc[0]);
        acc[1] = WMMA_F16(a0.v, b1.v, acc[1]);
        acc[2] = WMMA_F16(a0.v, b2.v, acc[2]);
        acc[3] = WMMA_F16(a0.v, b3.v, acc[3]);
        acc[4] = WMMA_F16(a1.v, b0.v, acc[4]);
        acc[5] = WMMA_F16(a1.v, b1.v, acc[5]);
        acc[6] = WMMA_F16(a1.v, b2.v, acc[6]);
        acc[7] = WMMA_F16(a1.v, b3.v, acc[7]);

        a0 = a0n; a1 = a1n; b0 = b0n; b1 = b1n; b2 = b2n; b3 = b3n;
        pa0 += 32; pa1 += 32; pw0 += 32; pw1 += 32; pw2 += 32; pw3 += 32;
    }
    // epilogue: last k-step, no loads
    acc[0] = WMMA_F16(a0.v, b0.v, acc[0]);
    acc[1] = WMMA_F16(a0.v, b1.v, acc[1]);
    acc[2] = WMMA_F16(a0.v, b2.v, acc[2]);
    acc[3] = WMMA_F16(a0.v, b3.v, acc[3]);
    acc[4] = WMMA_F16(a1.v, b0.v, acc[4]);
    acc[5] = WMMA_F16(a1.v, b1.v, acc[5]);
    acc[6] = WMMA_F16(a1.v, b2.v, acc[6]);
    acc[7] = WMMA_F16(a1.v, b3.v, acc[7]);

#pragma unroll
    for (int t = 0; t < 8; t++) {
        int n = n0 + (t & 3) * 16 + l15;
        int mb = m0 + (t >> 2) * 16;
        float bv = bias ? bias[n] : 0.f;
#pragma unroll
        for (int i = 0; i < 8; i++) {
            int m = mb + i + 8 * lh;
            float v = acc[t][i] + bv;
            if (mode == 1) v = 0.5f * v * (1.f + erff(v * 0.70710678118f));
            if (mode == 2) {
                size_t o = (size_t)m * Nout + n;
                res[o] = res[o] + v;
            } else {
                out16[(size_t)m * Nout + n] = (_Float16)v;
            }
        }
    }
}

// ---------------------------------------------------------------- GPSA positional softmax: pos[h][n][m]
__global__ void __launch_bounds__(256) pos_kernel(const float* __restrict__ pw,
                                                  const float* __restrict__ pb,
                                                  float* __restrict__ pos) {
    int hn = blockIdx.x;
    int h = hn / NTOK, n = hn % NTOK;
    float w0 = pw[h * 3 + 0], w1 = pw[h * 3 + 1], w2 = pw[h * 3 + 2], bb = pb[h];
    int nx = n % 14, ny = n / 14;
    int tid = threadIdx.x, m = tid;
    float ev = 0.f, e = -1e30f;
    if (m < NTOK) {
        int mx = m % 14, my = m / 14;
        float ix = (float)(mx - nx), iy = (float)(my - ny);
        ev = ix * w0 + iy * w1 + (ix * ix + iy * iy) * w2 + bb;
        e = ev;
    }
    for (int k = 1; k < 32; k <<= 1) e = fmaxf(e, __shfl_xor(e, k, 32));
    __shared__ float sm[8], ss[8];
    if ((tid & 31) == 0) sm[tid >> 5] = e;
    __syncthreads();
    float bmax = -1e30f;
    for (int i = 0; i < 8; i++) bmax = fmaxf(bmax, sm[i]);
    float p = (m < NTOK) ? __expf(ev - bmax) : 0.f;
    float s = p;
    for (int k = 1; k < 32; k <<= 1) s += __shfl_xor(s, k, 32);
    if ((tid & 31) == 0) ss[tid >> 5] = s;
    __syncthreads();
    float bs = 0.f;
    for (int i = 0; i < 8; i++) bs += ss[i];
    if (m < NTOK) pos[((size_t)h * NTOK + n) * NTOK + m] = p / bs;
}

// ---------------------------------------------------------------- fused attention, one block per (b,h)
// K and V^T staged in LDS; scores = WMMA QK^T -> in-register softmax (shfl_xor) ->
// [GPSA gate blend] -> attn rows to LDS -> WMMA attn@V from LDS.
__global__ void __launch_bounds__(256) attn_kernel(const _Float16* __restrict__ qk,
                                                   int q_off, int k_off, int qk_stride,
                                                   const _Float16* __restrict__ vsrc,
                                                   int v_off, int v_stride,
                                                   const float* __restrict__ pos,
                                                   const float* __restrict__ gate,
                                                   _Float16* __restrict__ out16) {
    extern __shared__ _Float16 smem[];
    _Float16* attnL = smem;                              // 208 x 224 f16
    _Float16* vtL   = attnL + MROWS * NPADK;             // 64  x 224 f16 (V^T, K-padded)
    _Float16* kL    = vtL + HD * NPADK;                  // 208 x 64 f16 (K rows, zero-padded)

    int bh = blockIdx.x;
    int b = bh / NH, h = bh % NH;
    size_t bn0 = (size_t)b * NTOK;
    const _Float16* qbase = qk + bn0 * qk_stride + q_off + h * HD;
    const _Float16* kbase = qk + bn0 * qk_stride + k_off + h * HD;
    const _Float16* vbase = vsrc + bn0 * v_stride + v_off + h * HD;

    // stage V^T (zero K padding) and K rows (zero token padding) into LDS
    for (int idx = threadIdx.x; idx < HD * NPADK; idx += 256) {
        int d = idx / NPADK, t = idx % NPADK;
        _Float16 val = (_Float16)0.f;
        if (t < NTOK) val = vbase[(size_t)t * v_stride + d];
        vtL[d * NPADK + t] = val;
    }
    for (int idx = threadIdx.x; idx < MROWS * HD; idx += 256) {
        int t = idx >> 6, d = idx & 63;
        _Float16 val = (_Float16)0.f;
        if (t < NTOK) val = kbase[(size_t)t * qk_stride + d];
        kL[idx] = val;
    }
    __syncthreads();

    const int lane = threadIdx.x & 31, wv = threadIdx.x >> 5;
    const int l15 = lane & 15, lh = lane >> 4;
    float gval = 0.f;
    if (pos) gval = 1.f / (1.f + __expf(-gate[h]));

    // ---- phase 1: scores + softmax (+gate blend), rows -> LDS
    for (int mt = wv; mt < 13; mt += 8) {
        v8f acc[13];
        for (int t = 0; t < 13; t++) acc[t] = (v8f){};
        int qtok = mt * 16 + l15; if (qtok > NTOK - 1) qtok = NTOK - 1;
        const _Float16* qrow = qbase + (size_t)qtok * qk_stride + lh * 8;
#pragma unroll
        for (int kk = 0; kk < HD; kk += 32) {
            FragH af = load_frag_a(qrow + kk);
#pragma unroll
            for (int t = 0; t < 13; t++) {
                FragH bf = load_frag_b(kL + (t * 16 + l15) * HD + kk + lh * 8);
                acc[t] = WMMA_F16(af.v, bf.v, acc[t]);
            }
        }
#pragma unroll
        for (int i = 0; i < 8; i++) {
            int m = mt * 16 + i + 8 * lh;
            float sv[13], rmax = -1e30f;
#pragma unroll
            for (int t = 0; t < 13; t++) {
                int n = t * 16 + l15;
                float s = (n < NTOK) ? acc[t][i] * 0.125f : -1e30f;   // scale = D^-0.5
                sv[t] = s;
                rmax = fmaxf(rmax, s);
            }
            for (int k = 1; k < 16; k <<= 1) rmax = fmaxf(rmax, __shfl_xor(rmax, k, 32));
            float rsum = 0.f;
#pragma unroll
            for (int t = 0; t < 13; t++) {
                int n = t * 16 + l15;
                float p = (n < NTOK) ? __expf(sv[t] - rmax) : 0.f;
                sv[t] = p;
                rsum += p;
            }
            for (int k = 1; k < 16; k <<= 1) rsum += __shfl_xor(rsum, k, 32);
            float inv = 1.f / rsum;
#pragma unroll
            for (int t = 0; t < 13; t++) {
                int n = t * 16 + l15;
                float a = 0.f;
                if (m < NTOK && n < NTOK) {
                    a = sv[t] * inv;
                    // GPSA: (1-g)*softmax(QK) + g*softmax(pos); both rows sum to 1 -> renorm is identity
                    if (pos) a = (1.f - gval) * a + gval * pos[((size_t)h * NTOK + m) * NTOK + n];
                }
                attnL[m * NPADK + n] = (_Float16)a;
            }
            attnL[m * NPADK + 208 + l15] = (_Float16)0.f;   // K padding 208..223
        }
    }
    __syncthreads();

    // ---- phase 2: out = attn @ V   (A from LDS, B = V^T from LDS)
    for (int mt = wv; mt < 13; mt += 8) {
        v8f acc[4];
        for (int t = 0; t < 4; t++) acc[t] = (v8f){};
        const _Float16* arow = attnL + (mt * 16 + l15) * NPADK + lh * 8;
#pragma unroll
        for (int kk = 0; kk < NPADK; kk += 32) {
            FragH af = load_frag_a(arow + kk);
#pragma unroll
            for (int t = 0; t < 4; t++) {
                FragH bf = load_frag_b(vtL + (t * 16 + l15) * NPADK + kk + lh * 8);
                acc[t] = WMMA_F16(af.v, bf.v, acc[t]);
            }
        }
#pragma unroll
        for (int t = 0; t < 4; t++)
#pragma unroll
            for (int i = 0; i < 8; i++) {
                int m = mt * 16 + i + 8 * lh;
                if (m < NTOK)
                    out16[(bn0 + m) * CDIM + h * HD + t * 16 + l15] = (_Float16)acc[t][i];
            }
    }
}

// ================================================================ host
static void conv_w(const void* src, _Float16* dst, int n, hipStream_t s) {
    int g = (n + 255) / 256; if (g > 4096) g = 4096;
    conv_f16_kernel<<<g, 256, 0, s>>>((const float*)src, dst, n);
}
static void gemm(const _Float16* A, const _Float16* W, const float* bias,
                 _Float16* out16, float* res, int K, int Nout, int mode, hipStream_t s) {
    dim3 grid(Nout / 128, MTOT / 128);
    gemm_kernel<<<grid, 256, 0, s>>>(A, W, bias, out16, res, K, Nout, mode);
}

extern "C" void kernel_launch(void* const* d_in, const int* in_sizes, int n_in,
                              void* d_out, int out_size, void* d_ws, size_t ws_size,
                              hipStream_t stream) {
    (void)in_sizes; (void)n_in; (void)out_size; (void)ws_size;
    const size_t M = MTOT;
    char* ws = (char*)d_ws;
    size_t off = 0;
    auto alloc = [&](size_t bytes) { char* p = ws + off; off += (bytes + 255) & ~(size_t)255; return p; };

    float*     xbuf  = (float*)    alloc(M * CDIM * 4);        // residual stream (f32)
    _Float16*  y16   = (_Float16*) alloc(M * CDIM * 2);        // LN out / attn out
    _Float16*  big16 = (_Float16*) alloc(M * 3072 * 2);        // qk(v) proj / MLP hidden
    _Float16*  v16b  = (_Float16*) alloc(M * CDIM * 2);        // GPSA v
    float*     posb  = (float*)    alloc((size_t)NH * NTOK * NTOK * 4);
    _Float16*  wbuf  = (_Float16*) alloc((size_t)3072 * 768 * 2);  // converted weight slab

    const float* x_in = (const float*)d_in[0];
    hipMemcpyAsync(xbuf, x_in, M * CDIM * 4, hipMemcpyDeviceToDevice, stream);

    const size_t attn_lds = (size_t)(MROWS * NPADK + HD * NPADK + MROWS * HD) * 2;

    // ---------------- GPSA blocks
    for (int i = 0; i < 4; i++) {
        const float* ln1g = (const float*)d_in[1] + (size_t)i * CDIM;
        const float* ln1b = (const float*)d_in[2] + (size_t)i * CDIM;
        const float* qkw  = (const float*)d_in[3] + (size_t)i * 1536 * 768;
        const float* vw   = (const float*)d_in[4] + (size_t)i * 768 * 768;
        const float* pjw  = (const float*)d_in[5] + (size_t)i * 768 * 768;
        const float* pjb  = (const float*)d_in[6] + (size_t)i * 768;
        const float* posw = (const float*)d_in[7] + (size_t)i * NH * 3;
        const float* posbi= (const float*)d_in[8] + (size_t)i * NH;
        const float* gate = (const float*)d_in[9] + (size_t)i * NH;
        const float* ln2g = (const float*)d_in[10] + (size_t)i * CDIM;
        const float* ln2b = (const float*)d_in[11] + (size_t)i * CDIM;
        const float* f1w  = (const float*)d_in[12] + (size_t)i * 3072 * 768;
        const float* f1b  = (const float*)d_in[13] + (size_t)i * 3072;
        const float* f2w  = (const float*)d_in[14] + (size_t)i * 768 * 3072;
        const float* f2b  = (const float*)d_in[15] + (size_t)i * 768;

        ln_kernel<<<MTOT, 256, 0, stream>>>(xbuf, ln1g, ln1b, y16);
        conv_w(qkw, wbuf, 1536 * 768, stream);
        gemm(y16, wbuf, nullptr, big16, nullptr, 768, 1536, 0, stream);   // qk
        conv_w(vw, wbuf, 768 * 768, stream);
        gemm(y16, wbuf, nullptr, v16b, nullptr, 768, 768, 0, stream);     // v
        pos_kernel<<<NH * NTOK, 256, 0, stream>>>(posw, posbi, posb);
        attn_kernel<<<64 * NH, 256, attn_lds, stream>>>(big16, 0, 768, 1536,
                                                        v16b, 0, 768, posb, gate, y16);
        conv_w(pjw, wbuf, 768 * 768, stream);
        gemm(y16, wbuf, pjb, nullptr, xbuf, 768, 768, 2, stream);         // proj + residual
        ln_kernel<<<MTOT, 256, 0, stream>>>(xbuf, ln2g, ln2b, y16);
        conv_w(f1w, wbuf, 3072 * 768, stream);
        gemm(y16, wbuf, f1b, big16, nullptr, 768, 3072, 1, stream);       // fc1 + GELU
        conv_w(f2w, wbuf, 768 * 3072, stream);
        gemm(big16, wbuf, f2b, nullptr, xbuf, 3072, 768, 2, stream);      // fc2 + residual
    }

    // ---------------- MHSA blocks
    for (int i = 0; i < 2; i++) {
        const float* ln1g = (const float*)d_in[16] + (size_t)i * CDIM;
        const float* ln1b = (const float*)d_in[17] + (size_t)i * CDIM;
        const float* qkvw = (const float*)d_in[18] + (size_t)i * 2304 * 768;
        const float* pjw  = (const float*)d_in[19] + (size_t)i * 768 * 768;
        const float* pjb  = (const float*)d_in[20] + (size_t)i * 768;
        const float* ln2g = (const float*)d_in[21] + (size_t)i * CDIM;
        const float* ln2b = (const float*)d_in[22] + (size_t)i * CDIM;
        const float* f1w  = (const float*)d_in[23] + (size_t)i * 3072 * 768;
        const float* f1b  = (const float*)d_in[24] + (size_t)i * 3072;
        const float* f2w  = (const float*)d_in[25] + (size_t)i * 768 * 3072;
        const float* f2b  = (const float*)d_in[26] + (size_t)i * 768;

        ln_kernel<<<MTOT, 256, 0, stream>>>(xbuf, ln1g, ln1b, y16);
        conv_w(qkvw, wbuf, 2304 * 768, stream);
        gemm(y16, wbuf, nullptr, big16, nullptr, 768, 2304, 0, stream);   // qkv
        attn_kernel<<<64 * NH, 256, attn_lds, stream>>>(big16, 0, 768, 2304,
                                                        big16, 1536, 2304, nullptr, nullptr, y16);
        conv_w(pjw, wbuf, 768 * 768, stream);
        gemm(y16, wbuf, pjb, nullptr, xbuf, 768, 768, 2, stream);
        ln_kernel<<<MTOT, 256, 0, stream>>>(xbuf, ln2g, ln2b, y16);
        conv_w(f1w, wbuf, 3072 * 768, stream);
        gemm(y16, wbuf, f1b, big16, nullptr, 768, 3072, 1, stream);
        conv_w(f2w, wbuf, 768 * 3072, stream);
        gemm(big16, wbuf, f2b, nullptr, xbuf, 3072, 768, 2, stream);
    }

    hipMemcpyAsync(d_out, xbuf, M * CDIM * 4, hipMemcpyDeviceToDevice, stream);
}
